// GATClassifier_84782654423395
// MI455X (gfx1250) — compile-verified, matrix operand order checked by reading, not run on hardware
//
#include <hip/hip_runtime.h>
#include <hip/hip_bf16.h>

// ---------------- problem constants (match reference) ----------------
#define NN   20000      // nodes
#define EE   100000     // edges
#define GG   64         // graphs
#define INF_ 63         // input feature dim
#define FF   1024       // 8 heads * 128 = feature width of every GAT layer
#define HEADS 8
#define HD   128
#define EPSBN 1e-5f
#define KMAX 1024
#define LDSPAD 8        // halves of row padding -> conflict-free ds_load_b128
#define NWT  4          // 16x16 N-tiles per wave (wave computes 16x64 strip)

typedef __attribute__((ext_vector_type(16))) _Float16 v16h;
typedef __attribute__((ext_vector_type(8)))  _Float16 v8h;
typedef __attribute__((ext_vector_type(8)))  float    v8f;

// ---------------- small utility kernels ----------------
__global__ void k_fill_f32(float* p, float v, int n) {
  int i = blockIdx.x * blockDim.x + threadIdx.x;
  if (i < n) p[i] = v;
}
__global__ void k_fill_u32(unsigned* p, unsigned v, int n) {
  int i = blockIdx.x * blockDim.x + threadIdx.x;
  if (i < n) p[i] = v;
}

// convert [M,Kin] f32 -> [M,Kpad] f16 (zero-pad K)
__global__ void k_cvt_pad_f16(const float* __restrict__ in, _Float16* __restrict__ out,
                              int Kin, int Kpad, int total) {
  int i = blockIdx.x * blockDim.x + threadIdx.x;
  if (i >= total) return;
  int r = i / Kpad, k = i % Kpad;
  out[i] = (k < Kin) ? (_Float16)in[(size_t)r * Kin + k] : (_Float16)0.0f;
}

// W [K,Nc] f32  ->  Wt [Nc,Kpad] f16 (transpose, zero-pad K)
__global__ void k_wt_f16(const float* __restrict__ w, _Float16* __restrict__ wt,
                         int K, int Kpad, int Nc, int total) {
  int i = blockIdx.x * blockDim.x + threadIdx.x;
  if (i >= total) return;
  int n = i / Kpad, k = i % Kpad;
  wt[i] = (k < K) ? (_Float16)w[(size_t)k * Nc + n] : (_Float16)0.0f;
}

// ---------------- WMMA GEMM: C[M,Nc] = A[M,K](f16) * Bt[Nc,K](f16)^T + bias ----------------
// block = 256 threads = 8 waves. Block computes a 16(M) x 512(N) tile:
//  - 16 x K strip of A is staged once in LDS (padded stride), shared by all 8 waves
//  - each wave computes 16x64 via 4 f32 accumulators, reusing its A fragment 4x
// grid = (M/16, Nc/512); K multiple of 32, K <= KMAX.
__global__ void k_gemm_wmma(const _Float16* __restrict__ A, const _Float16* __restrict__ Bt,
                            const float* __restrict__ bias, float* __restrict__ C,
                            int M, int K, int Nc) {
  __shared__ _Float16 As[16 * (KMAX + LDSPAD)];
  const int strideL = K + LDSPAD;

  int wave = threadIdx.x >> 5;
  int lane = threadIdx.x & 31;
  int tileM = blockIdx.x * 16;
  int nBase = blockIdx.y * (8 * 16 * NWT) + wave * (16 * NWT);

  // cooperative copy: A[tileM..tileM+16) x [0..K) -> LDS (8-half chunks, rows don't straddle)
  for (int c = threadIdx.x; c < 2 * K; c += 256) {      // 2*K chunks of 8 halves
    int pos = c * 8;
    int r = pos / K, k = pos % K;
    *(v8h*)(As + r * strideL + k) = *(const v8h*)(A + (size_t)(tileM + r) * K + k);
  }
  __syncthreads();

  int lm = lane & 15;          // row (A) / col (B) within tile
  int kb = (lane >> 4) * 8;    // K sub-block base per ISA 16-bit fragment layout

  const _Float16* lA = As + lm * strideL + kb;
  const _Float16* bp = Bt + (size_t)(nBase + lm) * K + kb;

  v8f acc[NWT] = {};
  union U { v16h v; v8h h[2]; };
  for (int k0 = 0; k0 < K; k0 += 32) {
    U a;
    a.h[0] = *(const v8h*)(lA + k0);        // K = kb .. kb+7
    a.h[1] = *(const v8h*)(lA + k0 + 16);   // K = kb+16 .. kb+23
#pragma unroll
    for (int t = 0; t < NWT; ++t) {
      U b;
      b.h[0] = *(const v8h*)(bp + (size_t)t * 16 * K + k0);
      b.h[1] = *(const v8h*)(bp + (size_t)t * 16 * K + k0 + 16);
      acc[t] = __builtin_amdgcn_wmma_f32_16x16x32_f16(
          /*neg_a=*/false, a.v, /*neg_b=*/false, b.v,
          /*c_mod=*/(short)0, acc[t], /*reuse_a=*/false, /*reuse_b=*/false);
    }
  }

  int row0 = tileM + (lane >> 4) * 8;
#pragma unroll
  for (int t = 0; t < NWT; ++t) {
    int col = nBase + t * 16 + lm;
    float bv = bias ? bias[col] : 0.0f;
#pragma unroll
    for (int r = 0; r < 8; ++r)
      C[(size_t)(row0 + r) * Nc + col] = acc[t][r] + bv;
  }
}

// ---------------- GATv2 edge kernels ----------------
// one wave per (edge, head): sc[e,h] = sum_d lrelu02(fs[src][h,d]+fd[dst][h,d]) * attn[h,d]
__global__ void k_edge_score(const float* __restrict__ fs, const float* __restrict__ fd,
                             const float* __restrict__ attn, const int* __restrict__ src,
                             const int* __restrict__ dst, float* __restrict__ sc, int P) {
  int wave = threadIdx.x >> 5, lane = threadIdx.x & 31;
  int p = blockIdx.x * 8 + wave;
  if (p >= P) return;
  int e = p >> 3, h = p & 7;
  const float* pfs = fs + (size_t)src[e] * FF + h * HD;
  const float* pfd = fd + (size_t)dst[e] * FF + h * HD;
  const float* pa  = attn + h * HD;
  float acc = 0.0f;
  for (int i = lane; i < HD; i += 32) {
    float v = pfs[i] + pfd[i];
    v = (v > 0.0f) ? v : 0.2f * v;           // GATv2 negative_slope = 0.2
    acc += v * pa[i];
  }
  for (int off = 16; off; off >>= 1) acc += __shfl_xor(acc, off, 32);
  if (lane == 0) sc[p] = acc;
}

__device__ inline unsigned enc_f32(float f) {
  unsigned u = __float_as_uint(f);
  return (u & 0x80000000u) ? ~u : (u | 0x80000000u);
}
__device__ inline float dec_f32(unsigned u) {
  return __uint_as_float((u & 0x80000000u) ? (u & 0x7FFFFFFFu) : ~u);
}

__global__ void k_segmax(const float* __restrict__ sc, const int* __restrict__ dst,
                         unsigned* __restrict__ mu, int P) {
  int p = blockIdx.x * blockDim.x + threadIdx.x;
  if (p >= P) return;
  int e = p >> 3, h = p & 7;
  atomicMax(&mu[(size_t)dst[e] * HEADS + h], enc_f32(sc[p]));
}

__global__ void k_finmax(float* m, int n) {   // decode in place; -inf (no in-edges) -> 0
  int i = blockIdx.x * blockDim.x + threadIdx.x;
  if (i >= n) return;
  float f = dec_f32(((const unsigned*)m)[i]);
  m[i] = isfinite(f) ? f : 0.0f;
}

__global__ void k_exden(float* __restrict__ sc, const float* __restrict__ m,
                        const int* __restrict__ dst, float* __restrict__ den, int P) {
  int p = blockIdx.x * blockDim.x + threadIdx.x;
  if (p >= P) return;
  int e = p >> 3, h = p & 7;
  float v = __expf(sc[p] - m[(size_t)dst[e] * HEADS + h]);
  sc[p] = v;
  atomicAdd(&den[(size_t)dst[e] * HEADS + h], v);
}

// one block per edge: out[dst] += alpha * fs[src]
__global__ void k_aggregate(const float* __restrict__ fs, const float* __restrict__ ex,
                            const float* __restrict__ den, const int* __restrict__ src,
                            const int* __restrict__ dst, float* __restrict__ out) {
  int e = blockIdx.x;
  int s = src[e], d = dst[e];
  for (int j = threadIdx.x; j < FF; j += blockDim.x) {
    int h = j >> 7;
    float a = ex[(size_t)e * HEADS + h] / den[(size_t)d * HEADS + h];
    atomicAdd(&out[(size_t)d * FF + j], a * fs[(size_t)s * FF + j]);
  }
}

// ---------------- BatchNorm ----------------
// partial sums over row chunks: grid (FF/256, CHUNKS); atomics into sum/sumsq
#define BNCHUNKS 32
__global__ void k_bnstats(const float* __restrict__ x, const float* __restrict__ bias,
                          float* __restrict__ sum, float* __restrict__ sumsq, int Nrows) {
  int c = blockIdx.x * blockDim.x + threadIdx.x;
  int rows = (Nrows + BNCHUNKS - 1) / BNCHUNKS;
  int r0 = blockIdx.y * rows;
  int r1 = min(r0 + rows, Nrows);
  float b = bias[c], s = 0.0f, q = 0.0f;
  for (int r = r0; r < r1; ++r) {
    float v = x[(size_t)r * FF + c] + b;
    s += v; q += v * v;
  }
  atomicAdd(&sum[c], s);
  atomicAdd(&sumsq[c], q);
}
__global__ void k_bnfin(const float* __restrict__ sum, const float* __restrict__ sumsq,
                        float* __restrict__ mean, float* __restrict__ var, int Nrows) {
  int c = blockIdx.x * blockDim.x + threadIdx.x;
  if (c >= FF) return;
  float mu = sum[c] / (float)Nrows;
  mean[c] = mu;
  var[c]  = sumsq[c] / (float)Nrows - mu * mu;
}

// fused: x = lrelu001(BN(x + bias)); also emit f16 copy for next WMMA GEMM
__global__ void k_bnapply(float* __restrict__ x, _Float16* __restrict__ xh,
                          const float* __restrict__ bias, const float* __restrict__ mean,
                          const float* __restrict__ var, const float* __restrict__ gamma,
                          const float* __restrict__ beta, int n) {
  int i = blockIdx.x * blockDim.x + threadIdx.x;
  if (i >= n) return;
  int c = i & (FF - 1);
  float v = x[i] + bias[c];
  v = (v - mean[c]) * rsqrtf(var[c] + EPSBN) * gamma[c] + beta[c];
  v = (v > 0.0f) ? v : 0.01f * v;
  x[i] = v;
  xh[i] = (_Float16)v;
}

// ---------------- pooling + head ----------------
__global__ void k_pool_cnt(const int* __restrict__ gid, float* __restrict__ cnt, int n) {
  int i = blockIdx.x * blockDim.x + threadIdx.x;
  if (i < n) atomicAdd(&cnt[gid[i]], 1.0f);
}
__global__ void k_pool_sum(const float* __restrict__ x, const int* __restrict__ gid,
                           float* __restrict__ hg, int n) {
  int i = blockIdx.x * blockDim.x + threadIdx.x;
  if (i >= n) return;
  int r = i >> 10, c = i & (FF - 1);
  atomicAdd(&hg[(size_t)gid[r] * FF + c], x[i]);
}
__global__ void k_pool_div(float* __restrict__ hg, const float* __restrict__ cnt, int n) {
  int i = blockIdx.x * blockDim.x + threadIdx.x;
  if (i < n) hg[i] /= fmaxf(cnt[i >> 10], 1.0f);
}

__global__ void k_lrelu_cvt(float* __restrict__ z, _Float16* __restrict__ zh, int n) {
  int i = blockIdx.x * blockDim.x + threadIdx.x;
  if (i >= n) return;
  float v = z[i];
  v = (v > 0.0f) ? v : 0.01f * v;
  z[i] = v;
  zh[i] = (_Float16)v;
}

// fc3: one wave per output element (G x 18, K=1024), f32 dot
__global__ void k_fc3(const float* __restrict__ z, const float* __restrict__ w,
                      const float* __restrict__ b, float* __restrict__ o,
                      int Gn, int K, int O) {
  int wave = threadIdx.x >> 5, lane = threadIdx.x & 31;
  int p = blockIdx.x * (blockDim.x >> 5) + wave;
  if (p >= Gn * O) return;
  int g = p / O, c = p % O;
  float acc = 0.0f;
  for (int k = lane; k < K; k += 32) acc += z[(size_t)g * K + k] * w[(size_t)k * O + c];
  for (int off = 16; off; off >>= 1) acc += __shfl_xor(acc, off, 32);
  if (lane == 0) o[p] = acc + b[c];
}

// ---------------- host orchestration ----------------
static inline size_t al256(size_t x) { return (x + 255) & ~(size_t)255; }
static inline int cdiv(int a, int b) { return (a + b - 1) / b; }

extern "C" void kernel_launch(void* const* d_in, const int* in_sizes, int n_in,
                              void* d_out, int out_size, void* d_ws, size_t ws_size,
                              hipStream_t stream) {
  (void)in_sizes; (void)n_in; (void)out_size; (void)ws_size;

  const float* h   = (const float*)d_in[0];
  const int*   src = (const int*)d_in[1];
  const int*   dst = (const int*)d_in[2];
  const int*   gid = (const int*)d_in[3];
  const float* fc1_w = (const float*)d_in[28]; const float* fc1_b = (const float*)d_in[29];
  const float* fc2_w = (const float*)d_in[30]; const float* fc2_b = (const float*)d_in[31];
  const float* fc3_w = (const float*)d_in[32]; const float* fc3_b = (const float*)d_in[33];
  float* out = (float*)d_out;

  // -------- workspace carve --------
  char* p = (char*)d_ws; size_t off = 0;
  auto take = [&](size_t bytes) { void* q = p + off; off = al256(off + bytes); return q; };
  float*    xf    = (float*)   take((size_t)NN * FF * 4);   // activations / aggregation output
  _Float16* xh    = (_Float16*)take((size_t)NN * FF * 2);   // f16 activations (GEMM A)
  float*    fs    = (float*)   take((size_t)NN * FF * 4);
  float*    fd    = (float*)   take((size_t)NN * FF * 4);
  _Float16* wtA   = (_Float16*)take((size_t)FF * FF * 2);   // transposed f16 weights (src / fc)
  _Float16* wtB   = (_Float16*)take((size_t)FF * FF * 2);   // transposed f16 weights (dst / fc)
  float*    sc    = (float*)   take((size_t)EE * HEADS * 4);
  float*    mbuf  = (float*)   take((size_t)NN * HEADS * 4);
  float*    den   = (float*)   take((size_t)NN * HEADS * 4);
  float*    bnsum = (float*)   take((size_t)FF * 4);
  float*    bnsq  = (float*)   take((size_t)FF * 4);
  float*    mean  = (float*)   take((size_t)FF * 4);
  float*    var   = (float*)   take((size_t)FF * 4);
  float*    hg    = (float*)   take((size_t)GG * FF * 4);
  float*    cnt   = (float*)   take((size_t)GG * 4);
  float*    z1    = (float*)   take((size_t)GG * FF * 4);
  float*    z2    = (float*)   take((size_t)GG * FF * 4);
  _Float16* zh    = (_Float16*)take((size_t)GG * FF * 2);

  const unsigned ENC_NEG_INF = ~0xFF800000u;  // enc_f32(-inf)

  // layer-1 input: h [N,63] f32 -> xh [N,64] f16 (pad)
  {
    int tot = NN * 64;
    k_cvt_pad_f16<<<cdiv(tot, 256), 256, 0, stream>>>(h, xh, INF_, 64, tot);
  }

  for (int l = 0; l < 3; ++l) {
    int base = 4 + l * 8;
    const float* w_src = (const float*)d_in[base + 0];
    const float* b_src = (const float*)d_in[base + 1];
    const float* w_dst = (const float*)d_in[base + 2];
    const float* b_dst = (const float*)d_in[base + 3];
    const float* attn  = (const float*)d_in[base + 4];
    const float* bias  = (const float*)d_in[base + 5];
    const float* gamma = (const float*)d_in[base + 6];
    const float* beta  = (const float*)d_in[base + 7];

    int Kin  = (l == 0) ? INF_ : FF;
    int Kpad = (l == 0) ? 64   : FF;

    // transpose + f16-convert weights
    {
      int tot = FF * Kpad;
      k_wt_f16<<<cdiv(tot, 256), 256, 0, stream>>>(w_src, wtA, Kin, Kpad, FF, tot);
      k_wt_f16<<<cdiv(tot, 256), 256, 0, stream>>>(w_dst, wtB, Kin, Kpad, FF, tot);
    }

    // fs = xh @ w_src + b_src ; fd = xh @ w_dst + b_dst   (WMMA, LDS-staged A)
    dim3 gg(NN / 16, FF / 512);
    k_gemm_wmma<<<gg, 256, 0, stream>>>(xh, wtA, b_src, fs, NN, Kpad, FF);
    k_gemm_wmma<<<gg, 256, 0, stream>>>(xh, wtB, b_dst, fd, NN, Kpad, FF);

    // attention logits
    const int P = EE * HEADS;
    k_edge_score<<<P / 8, 256, 0, stream>>>(fs, fd, attn, src, dst, sc, P);

    // segment softmax over in-edges of dst
    k_fill_u32<<<cdiv(NN * HEADS, 256), 256, 0, stream>>>((unsigned*)mbuf, ENC_NEG_INF, NN * HEADS);
    k_segmax<<<cdiv(P, 256), 256, 0, stream>>>(sc, dst, (unsigned*)mbuf, P);
    k_finmax<<<cdiv(NN * HEADS, 256), 256, 0, stream>>>(mbuf, NN * HEADS);
    k_fill_f32<<<cdiv(NN * HEADS, 256), 256, 0, stream>>>(den, 0.0f, NN * HEADS);
    k_exden<<<cdiv(P, 256), 256, 0, stream>>>(sc, mbuf, dst, den, P);

    // weighted aggregation into xf
    k_fill_f32<<<cdiv(NN * FF, 256), 256, 0, stream>>>(xf, 0.0f, NN * FF);
    k_aggregate<<<EE, 256, 0, stream>>>(fs, sc, den, src, dst, xf);

    // bias + BN + leaky-relu(0.01); emit f16 for next layer
    k_fill_f32<<<cdiv(FF, 256), 256, 0, stream>>>(bnsum, 0.0f, FF);
    k_fill_f32<<<cdiv(FF, 256), 256, 0, stream>>>(bnsq, 0.0f, FF);
    dim3 gbn(FF / 256, BNCHUNKS);
    k_bnstats<<<gbn, 256, 0, stream>>>(xf, bias, bnsum, bnsq, NN);
    k_bnfin<<<FF / 256, 256, 0, stream>>>(bnsum, bnsq, mean, var, NN);
    k_bnapply<<<cdiv(NN * FF, 256), 256, 0, stream>>>(xf, xh, bias, mean, var, gamma, beta, NN * FF);
  }

  // -------- mean pooling per graph --------
  k_fill_f32<<<cdiv(GG * FF, 256), 256, 0, stream>>>(hg, 0.0f, GG * FF);
  k_fill_f32<<<1, 256, 0, stream>>>(cnt, 0.0f, GG);
  k_pool_cnt<<<cdiv(NN, 256), 256, 0, stream>>>(gid, cnt, NN);
  k_pool_sum<<<cdiv(NN * FF, 256), 256, 0, stream>>>(xf, gid, hg, NN * FF);
  k_pool_div<<<cdiv(GG * FF, 256), 256, 0, stream>>>(hg, cnt, GG * FF);

  // -------- MLP head --------
  {
    int tot = GG * FF;
    k_cvt_pad_f16<<<cdiv(tot, 256), 256, 0, stream>>>(hg, zh, FF, FF, tot);      // hg -> f16
    int wtot = FF * FF;
    k_wt_f16<<<cdiv(wtot, 256), 256, 0, stream>>>(fc1_w, wtA, FF, FF, FF, wtot);
    dim3 gm(GG / 16, FF / 512);
    k_gemm_wmma<<<gm, 256, 0, stream>>>(zh, wtA, fc1_b, z1, GG, FF, FF);
    k_lrelu_cvt<<<cdiv(tot, 256), 256, 0, stream>>>(z1, zh, tot);

    k_wt_f16<<<cdiv(wtot, 256), 256, 0, stream>>>(fc2_w, wtB, FF, FF, FF, wtot);
    k_gemm_wmma<<<gm, 256, 0, stream>>>(zh, wtB, fc2_b, z2, GG, FF, FF);
    k_lrelu_cvt<<<cdiv(tot, 256), 256, 0, stream>>>(z2, zh, tot);

    const int O = 18, Ptot = GG * O;                  // 1152 outputs, wave each
    k_fc3<<<cdiv(Ptot, 8), 256, 0, stream>>>(z2, fc3_w, fc3_b, out, GG, FF, O);
  }
}
